// CSSMDeiT3Block_60043642798170
// MI455X (gfx1250) — matrix operand order repaired and checked by reading, not compiled, not run on hardware
//
#include <hip/hip_runtime.h>
#include <cmath>

typedef __bf16  bf16x16 __attribute__((ext_vector_type(16)));
typedef __bf16  bf16x8  __attribute__((ext_vector_type(8)));
typedef float   f32x8   __attribute__((ext_vector_type(8)));

#define NTOK 6272
#define CDIM 768
#define HID  3072

// Async global->LDS via inline asm (builtin's params are LangAS cuda_device
// pointers, not spellable from HIP source). Wait uses the probed builtin.
#if __has_builtin(__builtin_amdgcn_s_wait_asynccnt)
#define USE_ASYNC_LDS 1
#else
#define USE_ASYNC_LDS 0
#endif

__device__ __forceinline__ void async_load_b128(const void* gsrc, void* lds_dst) {
    unsigned lds = (unsigned)(unsigned long long)lds_dst;  // low 32 bits give the LDS address
    asm volatile("global_load_async_to_lds_b128 %0, %1, off"
                 :
                 : "v"(lds), "v"(gsrc)
                 : "memory");
}

// ---------------------------------------------------------------------------
// Weight convert + transpose: src (K x N, f32 row-major) -> dst (N x K, bf16)
// ---------------------------------------------------------------------------
__global__ __launch_bounds__(256) void wconv_kernel(const float* __restrict__ src,
                                                    __bf16* __restrict__ dst,
                                                    int K, int N) {
    size_t idx = (size_t)blockIdx.x * 256 + threadIdx.x;
    if (idx < (size_t)K * N) {
        int k = (int)(idx / N);
        int n = (int)(idx % N);
        dst[(size_t)n * K + k] = (__bf16)src[idx];
    }
}

// ---------------------------------------------------------------------------
// LayerNorm over C=768 per token, bf16 output. One block (256 thr) per token.
// ---------------------------------------------------------------------------
__global__ __launch_bounds__(256) void ln_bf16_kernel(const float* __restrict__ x,
                                                      const float* __restrict__ sc,
                                                      const float* __restrict__ bi,
                                                      __bf16* __restrict__ out) {
    const int t = threadIdx.x;
    const size_t base = (size_t)blockIdx.x * CDIM;
    float v0 = x[base + t], v1 = x[base + t + 256], v2 = x[base + t + 512];
    __shared__ float red[256];
    red[t] = v0 + v1 + v2;
    __syncthreads();
    for (int o = 128; o > 0; o >>= 1) { if (t < o) red[t] += red[t + o]; __syncthreads(); }
    float mu = red[0] * (1.0f / CDIM);
    __syncthreads();
    float d0 = v0 - mu, d1 = v1 - mu, d2 = v2 - mu;
    red[t] = d0 * d0 + d1 * d1 + d2 * d2;
    __syncthreads();
    for (int o = 128; o > 0; o >>= 1) { if (t < o) red[t] += red[t + o]; __syncthreads(); }
    float rstd = rsqrtf(red[0] * (1.0f / CDIM) + 1e-6f);
    out[base + t]       = (__bf16)(d0 * rstd * sc[t]       + bi[t]);
    out[base + t + 256] = (__bf16)(d1 * rstd * sc[t + 256] + bi[t + 256]);
    out[base + t + 512] = (__bf16)(d2 * rstd * sc[t + 512] + bi[t + 512]);
}

// ---------------------------------------------------------------------------
// Recurrence init: step 1 of the scan is analytically (hx,hy) = (u, 0).
// ---------------------------------------------------------------------------
__global__ __launch_bounds__(256) void init_h_kernel(const float* __restrict__ u,
                                                     __bf16* __restrict__ hx,
                                                     __bf16* __restrict__ hy, int n) {
    int idx = blockIdx.x * 256 + threadIdx.x;
    if (idx < n) {
        hx[idx] = (__bf16)u[idx];
        hy[idx] = (__bf16)0.0f;
    }
}

// ---------------------------------------------------------------------------
// Generic bf16 WMMA GEMM, 128x128 block tile, 8 waves x (32x64), K step 32.
//   out = A(MxK,bf16) * Bt(NxK,bf16)^T   (f32 accum)
// EPI 0: outF = acc + bias[n]
// EPI 1: gate step: g = sigmoid(acc+bias); hx' = g*(a*hx - b*hy) + u;
//        hy' = g*(b*hx + a*hy)  (A0 = hx_t, A1 = hy_t; Ksplit = CDIM)
// EPI 2: outF = res + p0[n] * (acc + bias[n])      (layerscale + residual)
// EPI 3: outB = bf16(gelu_tanh(acc + bias[n]))
// ---------------------------------------------------------------------------
template <int EPI>
__global__ __launch_bounds__(256) void gemm_bf16_wmma(
    const __bf16* __restrict__ A0, const __bf16* __restrict__ A1,
    const __bf16* __restrict__ Bt,
    int N, int K, int lda, int ldb, int Ksplit,
    const float* __restrict__ bias, const float* __restrict__ p0,
    const float* __restrict__ p1, const float* __restrict__ res,
    const float* __restrict__ uext,
    const __bf16* __restrict__ hxe, const __bf16* __restrict__ hye,
    float* __restrict__ outF, __bf16* __restrict__ outB,
    __bf16* __restrict__ outB2) {

    __shared__ __align__(16) __bf16 sA[2][128 * 32];
    __shared__ __align__(16) __bf16 sB[2][128 * 32];

    const int tid  = threadIdx.x;
    const int lane = tid & 31;
    const int w    = tid >> 5;
    const int wm   = (w >> 1) * 32;   // wave M offset (4 waves along M)
    const int wn   = (w & 1) * 64;    // wave N offset (2 waves along N)
    const int mBlock = blockIdx.y * 128;
    const int nBlock = blockIdx.x * 128;

    f32x8 acc[2][4];
#pragma unroll
    for (int i = 0; i < 2; ++i)
#pragma unroll
        for (int j = 0; j < 4; ++j)
            acc[i][j] = f32x8{0.f, 0.f, 0.f, 0.f, 0.f, 0.f, 0.f, 0.f};

    const int rA   = lane & 15;
    const int half = lane >> 4;
    const int steps = K / 32;

#if USE_ASYNC_LDS
    // ---- async global->LDS, double buffered ------------------------------
    auto issue_stage = [&](int st, int kk) {
#pragma unroll
        for (int i = 0; i < 2; ++i) {
            int c = tid + i * 256;
            int row = c >> 2, q = c & 3;
            int kg = kk + q * 8;
            const __bf16* s = A0 + (size_t)(mBlock + row) * lda + kg;
            if (EPI == 1 && kg >= Ksplit)
                s = A1 + (size_t)(mBlock + row) * lda + (kg - Ksplit);
            async_load_b128(s, &sA[st][row * 32 + q * 8]);
        }
#pragma unroll
        for (int i = 0; i < 2; ++i) {
            int c = tid + i * 256;
            int row = c >> 2, q = c & 3;
            const __bf16* s = Bt + (size_t)(nBlock + row) * ldb + kk + q * 8;
            async_load_b128(s, &sB[st][row * 32 + q * 8]);
        }
    };

    issue_stage(0, 0);
    for (int k = 0; k < steps; ++k) {
        const int cur = k & 1;
        __builtin_amdgcn_s_wait_asynccnt(0);
        __syncthreads();
        if (k + 1 < steps) issue_stage(cur ^ 1, (k + 1) * 32);

        bf16x16 af[2], bfr[4];
#pragma unroll
        for (int mt = 0; mt < 2; ++mt) {
            const bf16x8* p = reinterpret_cast<const bf16x8*>(
                &sA[cur][(wm + mt * 16 + rA) * 32]);
            bf16x8 lo = p[half];
            bf16x8 hi = p[2 + half];
            af[mt] = __builtin_shufflevector(lo, hi, 0, 1, 2, 3, 4, 5, 6, 7, 8,
                                             9, 10, 11, 12, 13, 14, 15);
        }
#pragma unroll
        for (int nt = 0; nt < 4; ++nt) {
            const bf16x8* p = reinterpret_cast<const bf16x8*>(
                &sB[cur][(wn + nt * 16 + rA) * 32]);
            bf16x8 lo = p[half];
            bf16x8 hi = p[2 + half];
            bfr[nt] = __builtin_shufflevector(lo, hi, 0, 1, 2, 3, 4, 5, 6, 7, 8,
                                              9, 10, 11, 12, 13, 14, 15);
        }
#pragma unroll
        for (int mt = 0; mt < 2; ++mt)
#pragma unroll
            for (int nt = 0; nt < 4; ++nt)
                acc[mt][nt] = __builtin_amdgcn_wmma_f32_16x16x32_bf16(
                    false, af[mt], false, bfr[nt], (short)0, acc[mt][nt], false,
                    false);
    }
#else
    // ---- fallback: synchronous staging, single buffer --------------------
    for (int k = 0; k < steps; ++k) {
        const int kk = k * 32;
#pragma unroll
        for (int i = 0; i < 2; ++i) {
            int c = tid + i * 256;
            int row = c >> 2, q = c & 3;
            int kg = kk + q * 8;
            const __bf16* s = A0 + (size_t)(mBlock + row) * lda + kg;
            if (EPI == 1 && kg >= Ksplit)
                s = A1 + (size_t)(mBlock + row) * lda + (kg - Ksplit);
            *reinterpret_cast<uint4*>(&sA[0][row * 32 + q * 8]) =
                *reinterpret_cast<const uint4*>(s);
        }
#pragma unroll
        for (int i = 0; i < 2; ++i) {
            int c = tid + i * 256;
            int row = c >> 2, q = c & 3;
            const __bf16* s = Bt + (size_t)(nBlock + row) * ldb + kk + q * 8;
            *reinterpret_cast<uint4*>(&sB[0][row * 32 + q * 8]) =
                *reinterpret_cast<const uint4*>(s);
            if (i == 0 && kk + 32 < K) __builtin_prefetch(s + 32, 0, 0);
        }
        __syncthreads();
        bf16x16 af[2], bfr[4];
#pragma unroll
        for (int mt = 0; mt < 2; ++mt) {
            const bf16x8* p = reinterpret_cast<const bf16x8*>(
                &sA[0][(wm + mt * 16 + rA) * 32]);
            bf16x8 lo = p[half];
            bf16x8 hi = p[2 + half];
            af[mt] = __builtin_shufflevector(lo, hi, 0, 1, 2, 3, 4, 5, 6, 7, 8,
                                             9, 10, 11, 12, 13, 14, 15);
        }
#pragma unroll
        for (int nt = 0; nt < 4; ++nt) {
            const bf16x8* p = reinterpret_cast<const bf16x8*>(
                &sB[0][(wn + nt * 16 + rA) * 32]);
            bf16x8 lo = p[half];
            bf16x8 hi = p[2 + half];
            bfr[nt] = __builtin_shufflevector(lo, hi, 0, 1, 2, 3, 4, 5, 6, 7, 8,
                                              9, 10, 11, 12, 13, 14, 15);
        }
#pragma unroll
        for (int mt = 0; mt < 2; ++mt)
#pragma unroll
            for (int nt = 0; nt < 4; ++nt)
                acc[mt][nt] = __builtin_amdgcn_wmma_f32_16x16x32_bf16(
                    false, af[mt], false, bfr[nt], (short)0, acc[mt][nt], false,
                    false);
        __syncthreads();
    }
#endif

    // ---- epilogue; C/D layout: lane n = lane&15, VGPR r -> M = r + 8*(lane>>4)
#pragma unroll
    for (int mt = 0; mt < 2; ++mt) {
#pragma unroll
        for (int nt = 0; nt < 4; ++nt) {
            const int nG = nBlock + wn + nt * 16 + rA;
            const int mBase = mBlock + wm + mt * 16 + half * 8;
            const float bn = bias[nG];
            if (EPI == 0) {
#pragma unroll
                for (int r = 0; r < 8; ++r)
                    outF[(size_t)(mBase + r) * N + nG] = acc[mt][nt][r] + bn;
            } else if (EPI == 1) {
                const float an = p0[nG], brn = p1[nG];
#pragma unroll
                for (int r = 0; r < 8; ++r) {
                    size_t idx = (size_t)(mBase + r) * N + nG;
                    float pre = acc[mt][nt][r] + bn;
                    float g = 1.0f / (1.0f + expf(-pre));
                    float hx = (float)hxe[idx];
                    float hy = (float)hye[idx];
                    float hxn = g * (an * hx - brn * hy) + uext[idx];
                    float hyn = g * (brn * hx + an * hy);
                    outB[idx]  = (__bf16)hxn;
                    outB2[idx] = (__bf16)hyn;
                }
            } else if (EPI == 2) {
                const float gm = p0[nG];
#pragma unroll
                for (int r = 0; r < 8; ++r) {
                    size_t idx = (size_t)(mBase + r) * N + nG;
                    outF[idx] = res[idx] + gm * (acc[mt][nt][r] + bn);
                }
            } else {  // EPI 3: gelu (tanh approx, jax.nn.gelu default)
#pragma unroll
                for (int r = 0; r < 8; ++r) {
                    size_t idx = (size_t)(mBase + r) * N + nG;
                    float v = acc[mt][nt][r] + bn;
                    float g = 0.5f * v *
                              (1.0f + tanhf(0.7978845608028654f *
                                            (v + 0.044715f * v * v * v)));
                    outB[idx] = (__bf16)g;
                }
            }
        }
    }
}

// ---------------------------------------------------------------------------
extern "C" void kernel_launch(void* const* d_in, const int* in_sizes, int n_in,
                              void* d_out, int out_size, void* d_ws,
                              size_t ws_size, hipStream_t stream) {
    const float* x       = (const float*)d_in[0];
    const float* ln1_s   = (const float*)d_in[1];
    const float* ln1_b   = (const float*)d_in[2];
    const float* W_in    = (const float*)d_in[3];
    const float* b_in    = (const float*)d_in[4];
    const float* W_gate  = (const float*)d_in[5];
    const float* b_gate  = (const float*)d_in[6];
    const float* a_decay = (const float*)d_in[7];
    const float* b_rot   = (const float*)d_in[8];
    const float* W_out   = (const float*)d_in[9];
    const float* b_out   = (const float*)d_in[10];
    const float* gamma1  = (const float*)d_in[11];
    const float* ln2_s   = (const float*)d_in[12];
    const float* ln2_b   = (const float*)d_in[13];
    const float* W1      = (const float*)d_in[14];
    const float* b1      = (const float*)d_in[15];
    const float* W2      = (const float*)d_in[16];
    const float* b2      = (const float*)d_in[17];
    const float* gamma2  = (const float*)d_in[18];

    char* ws = (char*)d_ws;
    size_t off = 0;
    auto alloc = [&](size_t bytes) {
        size_t o = off;
        off += (bytes + 255) & ~(size_t)255;
        return o;
    };
    const size_t NC = (size_t)NTOK * CDIM;
    __bf16* wt_in   = (__bf16*)(ws + alloc((size_t)CDIM * CDIM * 2));
    __bf16* wt_gate = (__bf16*)(ws + alloc((size_t)CDIM * 2 * CDIM * 2));
    __bf16* wt_out  = (__bf16*)(ws + alloc((size_t)CDIM * CDIM * 2));
    __bf16* wt1     = (__bf16*)(ws + alloc((size_t)HID * CDIM * 2));
    __bf16* wt2     = (__bf16*)(ws + alloc((size_t)CDIM * HID * 2));
    __bf16* xnb     = (__bf16*)(ws + alloc(NC * 2));
    float*  u       = (float*)(ws + alloc(NC * 4));
    __bf16* hxA     = (__bf16*)(ws + alloc(NC * 2));
    __bf16* hyA     = (__bf16*)(ws + alloc(NC * 2));
    __bf16* hxB     = (__bf16*)(ws + alloc(NC * 2));
    __bf16* hyB     = (__bf16*)(ws + alloc(NC * 2));
    float*  x1      = (float*)(ws + alloc(NC * 4));
    __bf16* hbuf    = (__bf16*)(ws + alloc((size_t)NTOK * HID * 2));

    auto cdiv = [](size_t a, size_t b) { return (unsigned)((a + b - 1) / b); };

    // 1) convert + transpose weights to bf16 (N x K)
    wconv_kernel<<<cdiv((size_t)CDIM * CDIM, 256), 256, 0, stream>>>(W_in, wt_in, CDIM, CDIM);
    wconv_kernel<<<cdiv((size_t)2 * CDIM * CDIM, 256), 256, 0, stream>>>(W_gate, wt_gate, 2 * CDIM, CDIM);
    wconv_kernel<<<cdiv((size_t)CDIM * CDIM, 256), 256, 0, stream>>>(W_out, wt_out, CDIM, CDIM);
    wconv_kernel<<<cdiv((size_t)CDIM * HID, 256), 256, 0, stream>>>(W1, wt1, CDIM, HID);
    wconv_kernel<<<cdiv((size_t)HID * CDIM, 256), 256, 0, stream>>>(W2, wt2, HID, CDIM);

    // 2) LN1 -> xnb (bf16)
    ln_bf16_kernel<<<NTOK, 256, 0, stream>>>(x, ln1_s, ln1_b, xnb);

    // 3) u = xnb @ W_in + b_in  (f32)
    gemm_bf16_wmma<0><<<dim3(CDIM / 128, NTOK / 128), 256, 0, stream>>>(
        xnb, nullptr, wt_in, CDIM, CDIM, CDIM, CDIM, CDIM, b_in, nullptr,
        nullptr, nullptr, nullptr, nullptr, nullptr, u, nullptr, nullptr);

    // 4) step 1 of the scan analytically: (hx, hy) = (u, 0)
    init_h_kernel<<<cdiv(NC, 256), 256, 0, stream>>>(u, hxA, hyA, (int)NC);

    // 5) steps 2..8: gate GEMM (K=1536, split hx|hy) + fused opponent update
    for (int s = 0; s < 7; ++s) {
        const __bf16* sx = (s & 1) ? hxB : hxA;
        const __bf16* sy = (s & 1) ? hyB : hyA;
        __bf16* dx = (s & 1) ? hxA : hxB;
        __bf16* dy = (s & 1) ? hyA : hyB;
        gemm_bf16_wmma<1><<<dim3(CDIM / 128, NTOK / 128), 256, 0, stream>>>(
            sx, sy, wt_gate, CDIM, 2 * CDIM, CDIM, 2 * CDIM, CDIM, b_gate,
            a_decay, b_rot, nullptr, u, sx, sy, nullptr, dx, dy);
    }
    // after 7 steps final hx is in hxB

    // 6) x1 = x + gamma1 * (hx @ W_out + b_out)
    gemm_bf16_wmma<2><<<dim3(CDIM / 128, NTOK / 128), 256, 0, stream>>>(
        hxB, nullptr, wt_out, CDIM, CDIM, CDIM, CDIM, CDIM, b_out, gamma1,
        nullptr, x, nullptr, nullptr, nullptr, x1, nullptr, nullptr);

    // 7) LN2 -> xnb (bf16, reuse)
    ln_bf16_kernel<<<NTOK, 256, 0, stream>>>(x1, ln2_s, ln2_b, xnb);

    // 8) h = gelu(xnb @ W1 + b1)  (bf16)
    gemm_bf16_wmma<3><<<dim3(HID / 128, NTOK / 128), 256, 0, stream>>>(
        xnb, nullptr, wt1, HID, CDIM, CDIM, CDIM, CDIM, b1, nullptr, nullptr,
        nullptr, nullptr, nullptr, nullptr, nullptr, hbuf, nullptr);

    // 9) out = x1 + gamma2 * (h @ W2 + b2)
    gemm_bf16_wmma<2><<<dim3(CDIM / 128, NTOK / 128), 256, 0, stream>>>(
        hbuf, nullptr, wt2, CDIM, HID, HID, HID, HID, b2, gamma2, nullptr, x1,
        nullptr, nullptr, nullptr, (float*)d_out, nullptr, nullptr);
}
